// Cross_Attention_30528627540441
// MI455X (gfx1250) — matrix operand order repaired and checked by reading, not compiled, not run on hardware
//
#include <hip/hip_runtime.h>
#include <hip/hip_bf16.h>

#define NPTS   32768
#define NVIEWS 8
#define DIM    256
#define ROWS   32        // points per block
#define THREADS 256      // 8 wave32

typedef __attribute__((ext_vector_type(16))) __bf16 v16bf;
typedef __attribute__((ext_vector_type(4)))  __bf16 v4bf;
typedef __attribute__((ext_vector_type(8)))  float  v8f;

// A-fragment (16x32 bf16, M x K) from a bf16 LDS tile (row-major, stride DIM).
// ISA layout: lane<16 holds K {0..7, 16..23} of row m=lane; lane>=16 holds K {8..15, 24..31}.
// Two contiguous 16B groups per lane -> 2x ds_load_b128, no ALU.
__device__ __forceinline__ v16bf load_A_bf(const __bf16* A, int lane, int ks) {
  const int m  = lane & 15;
  const int kb = ks * 32 + ((lane >> 4) << 3);
  const __bf16* p0 = A + m * DIM + kb;
  v16bf r;
  #pragma unroll
  for (int i = 0; i < 8; ++i) { r[i] = p0[i]; r[8 + i] = p0[16 + i]; }
  return r;
}

// B-fragment (32x16 bf16, K x N) where B[k,n] = W[n, ks*32 + k] (W bf16 row-major 256x256).
// ISA layout: lane<16 holds K {0..7, 16..23} of column n=lane&15; lane>=16 holds K {8..15, 24..31}.
// => each lane: two 8-bf16 (16B) groups from W row n.
__device__ __forceinline__ v16bf load_B_w(const __bf16* W, int lane, int ct, int ks) {
  const int n = ct * 16 + (lane & 15);
  const int k = ks * 32 + ((lane >> 4) << 3);
  const __bf16* p = W + n * DIM + k;
  v16bf r;
  #pragma unroll
  for (int i = 0; i < 8; ++i) { r[i] = p[i]; r[8 + i] = p[16 + i]; }
  return r;
}

__device__ __forceinline__ v8f bf16_wmma(v16bf a, v16bf b, v8f c) {
  return __builtin_amdgcn_wmma_f32_16x16x32_bf16(false, a, false, b, (short)0, c,
                                                 false, false);
}

// C(32x256) = A(32x256 bf16 LDS) @ B, B[k,n] = W[n,k] (bf16, row-major 256x256).
// 8 waves x 2 column tiles each; B fragment reused across the 2 M tiles.
// Epi(acc, mrow_base, col) consumes one 8-row accumulator column.
template <typename Epi>
__device__ __forceinline__ void wmma_gemm_32x256(const __bf16* __restrict__ Abf,
                                                 const __bf16* __restrict__ W,
                                                 Epi epi) {
  const int lane = threadIdx.x & 31;
  const int wave = threadIdx.x >> 5;
  #pragma unroll
  for (int cti = 0; cti < 2; ++cti) {
    const int ct = wave * 2 + cti;
    v8f acc0 = {};
    v8f acc1 = {};
    #pragma unroll
    for (int ks = 0; ks < 8; ++ks) {
      v16bf B  = load_B_w(W, lane, ct, ks);
      v16bf A0 = load_A_bf(Abf,            lane, ks);
      v16bf A1 = load_A_bf(Abf + 16 * DIM, lane, ks);
      acc0 = bf16_wmma(A0, B, acc0);
      acc1 = bf16_wmma(A1, B, acc1);
    }
    // C layout: VGPR j -> M = j (lanes<16) / j+8 (lanes>=16), N = lane&15
    const int col = ct * 16 + (lane & 15);
    const int mb  = (lane & 16) ? 8 : 0;
    epi(acc0, mb, col);
    epi(acc1, 16 + mb, col);
  }
}

// Pre-pass: convert Wq/Wv/Wo to bf16 as-is, Wk to bf16 transposed (so the
// qk = qp @ Wk GEMM's B-operand is contiguous along K).
__global__ void cvt_weights_kernel(const float* __restrict__ Wq, const float* __restrict__ Wk,
                                   const float* __restrict__ Wv, const float* __restrict__ Wo,
                                   __bf16* __restrict__ ws) {
  const int i = blockIdx.x * blockDim.x + threadIdx.x;     // 0 .. 65535
  __bf16* wq  = ws;
  __bf16* wkT = ws + DIM * DIM;
  __bf16* wv  = ws + 2 * DIM * DIM;
  __bf16* wo  = ws + 3 * DIM * DIM;
  wq[i] = (__bf16)Wq[i];
  wv[i] = (__bf16)Wv[i];
  wo[i] = (__bf16)Wo[i];
  const int r = i >> 8, c = i & 255;
  wkT[c * DIM + r] = (__bf16)Wk[i];
}

__global__ __launch_bounds__(THREADS)
void fused_crossattn_kernel(const float* __restrict__ q, const float* __restrict__ k,
                            const float* __restrict__ v,
                            const __bf16* __restrict__ wq, const __bf16* __restrict__ wkT,
                            const __bf16* __restrict__ wv, const __bf16* __restrict__ wo,
                            const float* __restrict__ bo, float* __restrict__ out) {
  // bf16 A-operand tiles (aliased by lifetime):
  __shared__ __align__(16) __bf16 sBfA[ROWS * DIM];  // q(bf16)  -> out3(bf16)
  __shared__ __align__(16) __bf16 sBfB[ROWS * DIM];  // qp(bf16) -> vbar(bf16)
  // f32 tiles:
  __shared__ __align__(16) float  sQP [ROWS * DIM];  // qp f32 (persists, residual)
  __shared__ __align__(16) float  sF  [ROWS * DIM];  // qk f32 -> y f32
  __shared__ float sAttn[ROWS * NVIEWS];
  __shared__ float sBo[DIM];

  const int tid  = threadIdx.x;
  const int lane = tid & 31;
  const int wave = tid >> 5;
  const int n0   = blockIdx.x * ROWS;

  sBo[tid] = bo[tid];
  // stage q tile (32x256) into LDS as bf16, coalesced float4 reads
  {
    const float4* q4 = (const float4*)(q + (size_t)n0 * DIM);
    for (int i = tid; i < ROWS * (DIM / 4); i += THREADS) {
      float4 t = q4[i];
      v4bf b;
      b[0] = (__bf16)t.x; b[1] = (__bf16)t.y; b[2] = (__bf16)t.z; b[3] = (__bf16)t.w;
      *(v4bf*)(sBfA + i * 4) = b;
    }
  }
  __syncthreads();

  // GEMM1: qp = q @ Wq^T   (store f32 for residual + bf16 for GEMM2's A)
  wmma_gemm_32x256(sBfA, wq, [&](const v8f& a, int mb, int col) {
    #pragma unroll
    for (int j = 0; j < 8; ++j) {
      sQP [(mb + j) * DIM + col] = a[j];
      sBfB[(mb + j) * DIM + col] = (__bf16)a[j];
    }
  });
  __syncthreads();

  // GEMM2: qk = qp @ Wk   (transposed-Wk bf16; f32 result for the score dots)
  wmma_gemm_32x256(sBfB, wkT, [&](const v8f& a, int mb, int col) {
    #pragma unroll
    for (int j = 0; j < 8; ++j) sF[(mb + j) * DIM + col] = a[j];
  });
  __syncthreads();

  // Scores: s[n,w] = (qk[n] . k[w, n0+n]) / 16.  Wave w handles view w, all 32 n.
  {
    const int w = wave;
    const float* kbase = k + ((size_t)w * NPTS + n0) * DIM;
    for (int n = 0; n < ROWS; ++n) {
      const float4* kr = (const float4*)(kbase + (size_t)n * DIM);
      const float4* qr = (const float4*)(sF + n * DIM);
      float acc = 0.0f;
      #pragma unroll
      for (int j = 0; j < 2; ++j) {
        float4 kv = kr[lane * 2 + j];
        float4 qv = qr[lane * 2 + j];
        acc += kv.x * qv.x + kv.y * qv.y + kv.z * qv.z + kv.w * qv.w;
      }
      #pragma unroll
      for (int off = 16; off > 0; off >>= 1) acc += __shfl_xor(acc, off, 32);
      if (lane == 0) sAttn[n * NVIEWS + w] = acc * 0.0625f;   // 1/sqrt(256)
    }
  }
  __syncthreads();

  // Softmax over 8 views, one thread per point
  if (tid < ROWS) {
    float sv[NVIEWS], m = -INFINITY;
    #pragma unroll
    for (int w = 0; w < NVIEWS; ++w) { sv[w] = sAttn[tid * NVIEWS + w]; m = fmaxf(m, sv[w]); }
    float sum = 0.0f;
    #pragma unroll
    for (int w = 0; w < NVIEWS; ++w) { sv[w] = __expf(sv[w] - m); sum += sv[w]; }
    const float inv = 1.0f / sum;
    #pragma unroll
    for (int w = 0; w < NVIEWS; ++w) sAttn[tid * NVIEWS + w] = sv[w] * inv;
  }
  __syncthreads();

  // vbar[n,:] = sum_w a[n,w] * v[w, n0+n, :]  -> bf16 LDS (A of GEMM3)
  {
    for (int i = tid; i < ROWS * (DIM / 4); i += THREADS) {
      const int n = i >> 6, c = i & 63;
      float4 acc = make_float4(0.f, 0.f, 0.f, 0.f);
      #pragma unroll
      for (int w = 0; w < NVIEWS; ++w) {
        const float aw = sAttn[n * NVIEWS + w];
        const float4 vv = *(const float4*)(v + (((size_t)w * NPTS + n0 + n) * DIM) + c * 4);
        acc.x += aw * vv.x; acc.y += aw * vv.y; acc.z += aw * vv.z; acc.w += aw * vv.w;
      }
      v4bf b;
      b[0] = (__bf16)acc.x; b[1] = (__bf16)acc.y; b[2] = (__bf16)acc.z; b[3] = (__bf16)acc.w;
      *(v4bf*)(sBfB + i * 4) = b;
    }
  }
  __syncthreads();

  // GEMM3: out3 = vbar @ Wv^T  -> bf16 only (A of GEMM4)
  wmma_gemm_32x256(sBfB, wv, [&](const v8f& a, int mb, int col) {
    #pragma unroll
    for (int j = 0; j < 8; ++j) sBfA[(mb + j) * DIM + col] = (__bf16)a[j];
  });
  __syncthreads();

  // GEMM4 + epilogue: y = gelu(out3 @ Wo^T + bo) + qp  -> sF (f32)
  wmma_gemm_32x256(sBfA, wo, [&](const v8f& a, int mb, int col) {
    const float bb = sBo[col];
    #pragma unroll
    for (int j = 0; j < 8; ++j) {
      const int m = mb + j;
      float x = a[j] + bb;
      float g = 0.5f * x * (1.0f + erff(x * 0.70710678118654752f));
      sF[m * DIM + col] = g + sQP[m * DIM + col];
    }
  });
  __syncthreads();

  // Write y replicated across the 8 views: out[(n0+n)*2048 + v*256 + d], float4 coalesced
  {
    float4* o4 = (float4*)(out + (size_t)n0 * NVIEWS * DIM);
    const float4* y4 = (const float4*)sF;
    for (int i = tid; i < ROWS * NVIEWS * (DIM / 4); i += THREADS) {
      const int n   = i >> 9;        // / 512
      const int rem = i & 511;
      const int vw  = rem >> 6;
      const int c   = rem & 63;
      o4[(size_t)n * (NVIEWS * DIM / 4) + vw * (DIM / 4) + c] = y4[n * (DIM / 4) + c];
    }
  }
}

extern "C" void kernel_launch(void* const* d_in, const int* in_sizes, int n_in,
                              void* d_out, int out_size, void* d_ws, size_t ws_size,
                              hipStream_t stream) {
  (void)in_sizes; (void)n_in; (void)out_size; (void)ws_size;
  const float* q  = (const float*)d_in[0];
  const float* k  = (const float*)d_in[1];
  const float* v  = (const float*)d_in[2];
  const float* Wq = (const float*)d_in[3];
  const float* Wk = (const float*)d_in[4];
  const float* Wv = (const float*)d_in[5];
  const float* Wo = (const float*)d_in[6];
  const float* bo = (const float*)d_in[7];
  float* out = (float*)d_out;
  __bf16* wsb = (__bf16*)d_ws;   // 4 * 256*256 bf16 = 512 KB

  // 1) weight conversion (Wk transposed)
  cvt_weights_kernel<<<(DIM * DIM) / THREADS, THREADS, 0, stream>>>(Wq, Wk, Wv, Wo, wsb);

  // 2) fused projection + view-attention + output MLP
  const __bf16* wq  = wsb;
  const __bf16* wkT = wsb + DIM * DIM;
  const __bf16* wv  = wsb + 2 * DIM * DIM;
  const __bf16* wo  = wsb + 3 * DIM * DIM;
  fused_crossattn_kernel<<<NPTS / ROWS, THREADS, 0, stream>>>(q, k, v, wq, wkT, wv, wo, bo, out);
}
